// TopologyMaskModule_41609643163936
// MI455X (gfx1250) — matrix operand order repaired and checked by reading, not compiled, not run on hardware
//
#include <hip/hip_runtime.h>
#include <hip/hip_bf16.h>
#include <math.h>

typedef __attribute__((ext_vector_type(16))) _Float16 v16h;
typedef __attribute__((ext_vector_type(8)))  _Float16 v8h;
typedef __attribute__((ext_vector_type(8)))  float    v8f;

#define GAMMA_C   (-0.1f)
#define ETA_C     (1.1f)
#define EPSCLIP_C (1e-7f)

#define D_     128   // embedding dim
#define H1_    128   // layer-1 width
#define H2_    64    // layer-2 width
#define HPITCH 136   // f16 row pitch in LDS (128 + 8 pad -> 272B rows, conflict-free TR reads)
#define WAVES  6     // waves per block (LDS budget: 6*2*4352B + scores < 64KB)

// d_ws layout (in _Float16 units):
//   w1f: 8 ks * 8 nt  fragments (32 lanes x 16 halfs) = 32768 @ 0
//   w2f: 4 ks * 4 nt  fragments                        =  8192 @ 32768
//   w3f: 2 ks * 1 nt  fragments (W3 padded to 64x16)   =  1024 @ 40960

// ---------------- weight fragment packing + loss init ----------------
__global__ void pack_weights_kernel(const float* __restrict__ W1,
                                    const float* __restrict__ W2,
                                    const float* __restrict__ W3,
                                    _Float16* __restrict__ wf,
                                    float* __restrict__ loss) {
  int tid = blockIdx.x * blockDim.x + threadIdx.x;
  if (tid == 0) *loss = 0.0f;
  if (tid < 32768) {
    // W1: [K=256][N=128] row-major -> B fragments, fragment id f = ks*8 + nt
    int f  = tid >> 9;
    int r  = tid & 511;
    int L  = r >> 4;            // lane
    int e  = r & 15;            // element within v16h
    int ks = f >> 3, nt = f & 7;
    int K  = ks * 32 + (e >> 3) * 16 + (L >> 4) * 8 + (e & 7);   // table order: K = 8h+e (e<8), 16+8h+e-8 (e>=8)
    int N  = nt * 16 + (L & 15);
    wf[tid] = (_Float16)W1[K * H1_ + N];
  } else if (tid < 32768 + 8192) {
    int t  = tid - 32768;
    int f  = t >> 9;
    int r  = t & 511;
    int L  = r >> 4, e = r & 15;
    int ks = f >> 2, nt = f & 3;
    int K  = ks * 32 + (e >> 3) * 16 + (L >> 4) * 8 + (e & 7);
    int N  = nt * 16 + (L & 15);
    wf[tid] = (_Float16)W2[K * H2_ + N];
  } else if (tid < 32768 + 8192 + 1024) {
    int t  = tid - (32768 + 8192);
    int f  = t >> 9;            // ks 0..1
    int r  = t & 511;
    int L  = r >> 4, e = r & 15;
    int K  = f * 32 + (e >> 3) * 16 + (L >> 4) * 8 + (e & 7);
    int N  = L & 15;
    wf[tid] = (N == 0) ? (_Float16)W3[K] : (_Float16)0.0f;
  }
}

// ---------------- main fused kernel ----------------
union AF { v16h v; _Float16 h[16]; };

__device__ __forceinline__ void load_chunk8(const float* __restrict__ p, _Float16* dst) {
  const float4* p4 = (const float4*)p;       // 32B aligned (offset multiple of 8 f32)
  float4 c0 = p4[0];
  float4 c1 = p4[1];
  dst[0] = (_Float16)c0.x; dst[1] = (_Float16)c0.y;
  dst[2] = (_Float16)c0.z; dst[3] = (_Float16)c0.w;
  dst[4] = (_Float16)c1.x; dst[5] = (_Float16)c1.y;
  dst[6] = (_Float16)c1.z; dst[7] = (_Float16)c1.w;
}

__device__ __forceinline__ void lds_fence() {
  asm volatile("s_wait_dscnt 0" ::: "memory");
}

#define WMMA_F16(A, B, C) \
  __builtin_amdgcn_wmma_f32_16x16x32_f16(false, (A), false, (B), (short)0, (C), false, false)

__global__ __launch_bounds__(WAVES * 32)
void edge_mlp_kernel(const float* __restrict__ emb,
                     const int*   __restrict__ eidx,
                     const float* __restrict__ tptr,
                     const float* __restrict__ eps,
                     const float* __restrict__ b1,
                     const float* __restrict__ b2,
                     const float* __restrict__ b3,
                     const _Float16* __restrict__ wf,
                     float* __restrict__ masks,
                     float* __restrict__ scores_out,
                     float* __restrict__ loss,
                     int E) {
  // per-wave, per-M-tile activation transpose buffers + per-wave score slots
  __shared__ __align__(16) _Float16 hbuf[WAVES][2][16 * HPITCH];
  __shared__ float sscore[WAVES][32];

  const int  lane  = threadIdx.x & 31;
  const int  wave  = threadIdx.x >> 5;
  const long tbase = ((long)blockIdx.x * WAVES + wave) * 32;   // 32 edges per wave
  if (tbase >= (long)E) return;            // wave-uniform exit (no block barriers used)

  const int  half = lane >> 4;             // K-half for A/B fragments
  const int  m    = lane & 15;             // A row fed by this lane / C column
  const int  mrow = half * 8;              // C-layout row base for this lane half
  const long eA   = tbase + m;             // edge for M-tile 0
  const long eB   = tbase + 16 + m;        // edge for M-tile 1
  const long eAL  = (eA < (long)E) ? eA : (long)E - 1;
  const long eBL  = (eB < (long)E) ? eB : (long)E - 1;

  const float* srow0 = emb + (long)eidx[eAL] * D_;
  const float* trow0 = emb + ((long)eidx[(long)E + eAL]) * D_;
  const float* srow1 = emb + (long)eidx[eBL] * D_;
  const float* trow1 = emb + ((long)eidx[(long)E + eBL]) * D_;

  const v16h* w1f = (const v16h*)(wf);
  const v16h* w2f = (const v16h*)(wf + 32768);
  const v16h* w3f = (const v16h*)(wf + 32768 + 8192);

  // ---- gather A fragments for layer 1 (two 16x256 tiles, f32 -> f16) ----
  AF a1[2][8];
#pragma unroll
  for (int ks = 0; ks < 8; ++ks) {
    int o0 = (ks * 32 + half * 8) & 127;          // elems 0..7 : K = ks*32 + half*8 + j
    int o1 = (ks * 32 + 16 + half * 8) & 127;     // elems 8..15: K = ks*32+16+half*8+j
    const float* b0 = (ks < 4) ? srow0 : trow0;
    const float* b1p = (ks < 4) ? srow1 : trow1;
    load_chunk8(b0 + o0,  &a1[0][ks].h[0]);
    load_chunk8(b0 + o1,  &a1[0][ks].h[8]);
    load_chunk8(b1p + o0, &a1[1][ks].h[0]);
    load_chunk8(b1p + o1, &a1[1][ks].h[8]);
  }

  // ---- layer 1: h1 = relu(X(32x256) * W1(256x128) + b1) ----
  // B fragments double-buffered; each fragment feeds both M-tiles.
  v16h bq = w1f[lane];                       // f = ks*8+nt, first is (ks=0,nt=0) -> 0
#pragma unroll
  for (int nt = 0; nt < 8; ++nt) {
    v8f acc0 = {}, acc1 = {};
#pragma unroll
    for (int ks = 0; ks < 8; ++ks) {
      v16h b = bq;
      int fn = (ks < 7) ? ((ks + 1) * 8 + nt) : ((nt < 7) ? (nt + 1) : 0);
      bq = w1f[fn * 32 + lane];              // prefetch next fragment
      acc0 = WMMA_F16(a1[0][ks].v, b, acc0);
      acc1 = WMMA_F16(a1[1][ks].v, b, acc1);
    }
    int col  = nt * 16 + m;
    float bi = b1[col];
#pragma unroll
    for (int v = 0; v < 8; ++v) {
      float x0 = acc0[v] + bi;
      float x1 = acc1[v] + bi;
      hbuf[wave][0][(mrow + v) * HPITCH + col] = (_Float16)fmaxf(x0, 0.0f);
      hbuf[wave][1][(mrow + v) * HPITCH + col] = (_Float16)fmaxf(x1, 0.0f);
    }
  }
  lds_fence();

  // ---- transpose h1 (C layout) -> A fragments via per-wave LDS ----
  AF a2[2][4];
#pragma unroll
  for (int t = 0; t < 2; ++t) {
    const _Float16* rp = &hbuf[wave][t][m * HPITCH];
#pragma unroll
    for (int ks = 0; ks < 4; ++ks) {
      *(v8h*)&a2[t][ks].h[0] = *(const v8h*)(rp + ks * 32 + half * 8);
      *(v8h*)&a2[t][ks].h[8] = *(const v8h*)(rp + ks * 32 + 16 + half * 8);
    }
  }
  lds_fence();

  // ---- layer 2: h2 = relu(h1(32x128) * W2(128x64) + b2) ----
  v16h bq2 = w2f[lane];
#pragma unroll
  for (int nt = 0; nt < 4; ++nt) {
    v8f acc0 = {}, acc1 = {};
#pragma unroll
    for (int ks = 0; ks < 4; ++ks) {
      v16h b = bq2;
      int fn = (ks < 3) ? ((ks + 1) * 4 + nt) : ((nt < 3) ? (nt + 1) : 0);
      bq2 = w2f[fn * 32 + lane];
      acc0 = WMMA_F16(a2[0][ks].v, b, acc0);
      acc1 = WMMA_F16(a2[1][ks].v, b, acc1);
    }
    int col  = nt * 16 + m;
    float bi = b2[col];
#pragma unroll
    for (int v = 0; v < 8; ++v) {
      float x0 = acc0[v] + bi;
      float x1 = acc1[v] + bi;
      hbuf[wave][0][(mrow + v) * HPITCH + col] = (_Float16)fmaxf(x0, 0.0f);
      hbuf[wave][1][(mrow + v) * HPITCH + col] = (_Float16)fmaxf(x1, 0.0f);
    }
  }
  lds_fence();

  // ---- layer 3: scores = h2(32x64) * W3(64x16 padded, col 0 live) ----
  AF a3[2][2];
#pragma unroll
  for (int t = 0; t < 2; ++t) {
    const _Float16* rp = &hbuf[wave][t][m * HPITCH];
#pragma unroll
    for (int ks = 0; ks < 2; ++ks) {
      *(v8h*)&a3[t][ks].h[0] = *(const v8h*)(rp + ks * 32 + half * 8);
      *(v8h*)&a3[t][ks].h[8] = *(const v8h*)(rp + ks * 32 + 16 + half * 8);
    }
  }
  lds_fence();

  v16h w3a = w3f[lane];
  v16h w3b = w3f[32 + lane];
  v8f acc3a = {}, acc3b = {};
  acc3a = WMMA_F16(a3[0][0].v, w3a, acc3a);
  acc3a = WMMA_F16(a3[0][1].v, w3b, acc3a);
  acc3b = WMMA_F16(a3[1][0].v, w3a, acc3b);
  acc3b = WMMA_F16(a3[1][1].v, w3b, acc3b);

  // column N=0 lives in lanes 0 (M=0..7) and 16 (M=8..15)
  if (m == 0) {
#pragma unroll
    for (int v = 0; v < 8; ++v) {
      sscore[wave][mrow + v]      = acc3a[v];
      sscore[wave][16 + mrow + v] = acc3b[v];
    }
  }
  lds_fence();

  // ---- per-edge epilogue: all 32 lanes, one edge each (coalesced stores) ----
  float expected = 0.0f;
  const long e = tbase + lane;
  if (e < (long)E) {
    float sc  = sscore[wave][lane] + b3[0];
    float T   = tptr[0];
    float ec  = fminf(fmaxf(eps[e], EPSCLIP_C), 1.0f - EPSCLIP_C);
    float gum = logf(ec) - log1pf(-ec);
    float s   = 1.0f / (1.0f + expf(-((gum + sc) / T)));
    float msk = fminf(fmaxf(s * (ETA_C - GAMMA_C) + GAMMA_C, 0.0f), 1.0f);
    float se  = 1.0f / (1.0f + expf(-(sc / T)));
    expected  = fminf(fmaxf(se * (ETA_C - GAMMA_C) + GAMMA_C, 0.0f), 1.0f);
    masks[e]      = msk;
    scores_out[e] = sc;
  }
#pragma unroll
  for (int off = 16; off > 0; off >>= 1)
    expected += __shfl_xor(expected, off, 32);
  if (lane == 0) atomicAdd(loss, expected);
}

// ---------------- host launcher ----------------
extern "C" void kernel_launch(void* const* d_in, const int* in_sizes, int n_in,
                              void* d_out, int out_size, void* d_ws, size_t ws_size,
                              hipStream_t stream) {
  const float* emb  = (const float*)d_in[0];   // [N,128]
  const int*   eidx = (const int*)  d_in[1];   // [2,E] int32
  const float* tptr = (const float*)d_in[2];   // temperature scalar
  const float* eps  = (const float*)d_in[3];   // [E]
  const float* W1   = (const float*)d_in[4];   // [256,128]
  const float* b1   = (const float*)d_in[5];   // [128]
  const float* W2   = (const float*)d_in[6];   // [128,64]
  const float* b2   = (const float*)d_in[7];   // [64]
  const float* W3   = (const float*)d_in[8];   // [64,1]
  const float* b3   = (const float*)d_in[9];   // [1]

  const int E = in_sizes[3];

  _Float16* wf    = (_Float16*)d_ws;
  float*    out   = (float*)d_out;
  float*    masks = out;
  float*    sco   = out + E;
  float*    loss  = out + 2LL * E;

  const int packN = 32768 + 8192 + 1024;
  pack_weights_kernel<<<(packN + 255) / 256, 256, 0, stream>>>(W1, W2, W3, wf, loss);

  const long edgesPerBlock = (long)WAVES * 32;
  const int  blocks = (int)((E + edgesPerBlock - 1) / edgesPerBlock);
  edge_mlp_kernel<<<blocks, WAVES * 32, 0, stream>>>(emb, eidx, tptr, eps,
                                                     b1, b2, b3, wf,
                                                     masks, sco, loss, E);
}
